// VAE_74594991997710
// MI455X (gfx1250) — compile-verified, matrix operand order checked by reading, not run on hardware
//
#include <hip/hip_runtime.h>
#include <hip/hip_bf16.h>

// ---------- types ----------
typedef __attribute__((ext_vector_type(16))) __bf16 v16bf;
typedef __attribute__((ext_vector_type(8)))  float  v8f;

#define BTOT   65536
#define K1PAD  544      // 528 padded to 17*32
#define H0STR  560      // LDS row stride for h0 (pad)
#define N1     2048
#define NLAT   128
#define N3     1024
#define N4     264
#define N4PAD  272

// smem carve (bytes)
#define OFF_H0 0
#define OFF_H1 17920                    // 16*560*2
#define OFF_Z  (17920 + 65536)          // + 16*2048*2
#define OFF_H3 (17920 + 65536 + 4096)   // + 16*128*2
#define SMEM_BYTES (17920 + 65536 + 4096 + 32768)  // 120320

__device__ __forceinline__ unsigned short f2bf(float f) {
  unsigned int u = __float_as_uint(f);
  u += 0x7fffu + ((u >> 16) & 1u);      // round-to-nearest-even
  return (unsigned short)(u >> 16);
}

__device__ __forceinline__ float selu_f(float x) {
  const float scale = 1.0507009873554805f;
  const float alpha = 1.6732632423543772f;
  return scale * (x > 0.f ? x : alpha * (__expf(x) - 1.f));
}

union FragBF {
  v16bf v;
  uint4 q[2];
};

// Load one 16x32 bf16 operand fragment (A rows or W rows; both K-contiguous).
// CDNA5 16-bit operand layout: lane L holds row (L&15), K-chunks
// [k0+hi .. +7] and [k0+hi+16 .. +23], hi = (L>=16 ? 8 : 0).
__device__ __forceinline__ v16bf load_frag(const unsigned short* base,
                                           int rowStride, int k0, int lane) {
  const unsigned short* p = base + (lane & 15) * rowStride + k0 + ((lane >> 4) << 3);
  FragBF f;
  f.q[0] = *(const uint4*)(p);
  f.q[1] = *(const uint4*)(p + 16);
  return f.v;
}

#define WMMA_BF16(A, Bm, C) \
  __builtin_amdgcn_wmma_f32_16x16x32_bf16(false, (A), false, (Bm), (short)0, (C), false, false)

// ---------- weight convert + pad to bf16 (dst: Nd x Kd, src: N x Ks) ----------
__global__ void cvt_pad(const float* __restrict__ src, unsigned short* __restrict__ dst,
                        int N, int Ks, int Kd, int Nd) {
  int i = blockIdx.x * blockDim.x + threadIdx.x;
  int tot = Nd * Kd;
  int stride = gridDim.x * blockDim.x;
  for (; i < tot; i += stride) {
    int n = i / Kd, k = i - n * Kd;
    float v = (n < N && k < Ks) ? src[n * Ks + k] : 0.f;
    dst[i] = f2bf(v);
  }
}

// ---------- fully fused VAE forward: 16 batch rows per workgroup ----------
__global__ void __launch_bounds__(256, 1)
vae_fused(const float* __restrict__ x, const float* __restrict__ eps,
          const unsigned short* __restrict__ W1b,  const float* __restrict__ b1,
          const unsigned short* __restrict__ W21b, const float* __restrict__ b21,
          const unsigned short* __restrict__ W22b, const float* __restrict__ b22,
          const unsigned short* __restrict__ W3b,  const float* __restrict__ b3,
          const unsigned short* __restrict__ W4b,  const float* __restrict__ b4,
          float* __restrict__ out) {
  extern __shared__ __align__(16) unsigned char smem[];
  unsigned short* sh0 = (unsigned short*)(smem + OFF_H0);  // 16 x 560 (K=544 used)
  unsigned short* sh1 = (unsigned short*)(smem + OFF_H1);  // 16 x 2048
  unsigned short* shz = (unsigned short*)(smem + OFF_Z);   // 16 x 128
  unsigned short* sh3 = (unsigned short*)(smem + OFF_H3);  // 16 x 1024
  float* shx = (float*)(smem + OFF_H1);                    // overlay: 16 x 264 f32

  const int tid  = threadIdx.x;
  const int wave = tid >> 5;
  const int lane = tid & 31;
  const int r0   = blockIdx.x << 4;   // first batch row of this block
  const int hi   = (lane >> 4) << 3;  // C/D row offset for this half-wave
  const int cl   = lane & 15;         // C/D column within tile

  // ---- prologue: stage x rows (contiguous) into LDS ----
  for (int i = tid; i < 16 * 264; i += 256)
    shx[i] = x[(size_t)r0 * 264 + i];
  __syncthreads();

  // ---- feature engineering -> h0 (bf16, padded to 544) ----
  for (int i = tid; i < 16 * K1PAD; i += 256) {
    int row = i / K1PAD;
    int j   = i - row * K1PAD;
    const float* xr = shx + row * 264;
    float v;
    if (j < 264) {                       // stateVec
      v = xr[j];
    } else if (j < 396) {                // thickness
      int t = j - 264;
      v = (t & 1) ? fabsf(xr[t] - xr[t + 132]) : xr[t];
    } else if (j < 528) {                // angles
      int a    = j - 396;
      int ip1  = (a + 1) % 132;
      int i2   = ((a + 1) % 66) * 2;
      int i2p1 = (i2 + 1) % 132;
      float v1x = xr[2 * ip1]     - xr[2 * a];
      float v1y = xr[2 * ip1 + 1] - xr[2 * a + 1];
      float v2x = xr[2 * i2p1]     - xr[2 * i2];
      float v2y = xr[2 * i2p1 + 1] - xr[2 * i2 + 1];
      float c = (v1x * v2x + v1y * v2y) *
                rsqrtf((v1x * v1x + v1y * v1y) * (v2x * v2x + v2y * v2y));
      c = fminf(fmaxf(c, -1.f + 1e-6f), 1.f - 1e-6f);
      v = acosf(c);
    } else {                             // zero pad 528..543
      v = 0.f;
    }
    sh0[row * H0STR + j] = f2bf(v);
  }
  __syncthreads();

  // ---- Stage 1: h1 = selu(h0 @ W1^T + b1)  [16 x 2048, K=544] ----
  // 4 N-tiles per K-pass: 1 shared A fragment -> 4 WMMAs.
  {
    for (int g = 0; g < 4; ++g) {
      const int n0 = wave * 256 + g * 64;
      const unsigned short* wb[4];
#pragma unroll
      for (int j = 0; j < 4; ++j) wb[j] = W1b + (size_t)(n0 + 16 * j) * K1PAD;
      v8f acc[4] = {};
      for (int kb = 0; kb < K1PAD / 32; ++kb) {
        const int k0 = kb * 32;
        v16bf a = load_frag(sh0, H0STR, k0, lane);
#pragma unroll
        for (int j = 0; j < 4; ++j) {
          __builtin_prefetch(wb[j] + k0 + 64, 0, 3);
          v16bf b = load_frag(wb[j], K1PAD, k0, lane);
          acc[j] = WMMA_BF16(a, b, acc[j]);
        }
      }
#pragma unroll
      for (int j = 0; j < 4; ++j) {
        const int col = n0 + j * 16 + cl;
        const float bias = b1[col];
#pragma unroll
        for (int r = 0; r < 8; ++r)
          sh1[(r + hi) * N1 + col] = f2bf(selu_f(acc[j][r] + bias));
      }
    }
  }
  __syncthreads();

  // ---- Stage 2: mu/logvar [16 x 128 each, K=2048]; shared A fragment ----
  {
    const int n0 = wave * 16;            // 8 waves cover 128 cols for both heads
    const unsigned short* wMu = W21b + (size_t)n0 * N1;
    const unsigned short* wLv = W22b + (size_t)n0 * N1;
    v8f aMu = {};
    v8f aLv = {};
    for (int kb = 0; kb < N1 / 32; ++kb) {
      const int k0 = kb * 32;
      __builtin_prefetch(wMu + k0 + 64, 0, 3);
      __builtin_prefetch(wLv + k0 + 64, 0, 3);
      v16bf a  = load_frag(sh1, N1, k0, lane);
      v16bf bm = load_frag(wMu, N1, k0, lane);
      aMu = WMMA_BF16(a, bm, aMu);
      v16bf bl = load_frag(wLv, N1, k0, lane);
      aLv = WMMA_BF16(a, bl, aLv);
    }
    const int col = n0 + cl;
    const float bm = b21[col], bl = b22[col];
    const size_t muBase = (size_t)BTOT * 264;
    const size_t lvBase = muBase + (size_t)BTOT * 128;
#pragma unroll
    for (int r = 0; r < 8; ++r) {
      int m = r + hi;
      size_t grow = (size_t)(r0 + m);
      float mu = aMu[r] + bm;
      float lv = aLv[r] + bl;
      float z  = mu + eps[grow * 128 + col] * __expf(0.5f * lv);
      out[muBase + grow * 128 + col] = mu;
      out[lvBase + grow * 128 + col] = lv;
      shz[m * NLAT + col] = f2bf(z);
    }
  }
  __syncthreads();

  // ---- Stage 3: h3 = selu(z @ W3^T + b3)  [16 x 1024, K=128]; 4 tiles/pass ----
  {
    for (int g = 0; g < 2; ++g) {
      const int n0 = wave * 128 + g * 64;
      const unsigned short* wb[4];
#pragma unroll
      for (int j = 0; j < 4; ++j) wb[j] = W3b + (size_t)(n0 + 16 * j) * NLAT;
      v8f acc[4] = {};
#pragma unroll
      for (int kb = 0; kb < NLAT / 32; ++kb) {
        const int k0 = kb * 32;
        v16bf a = load_frag(shz, NLAT, k0, lane);
#pragma unroll
        for (int j = 0; j < 4; ++j) {
          v16bf b = load_frag(wb[j], NLAT, k0, lane);
          acc[j] = WMMA_BF16(a, b, acc[j]);
        }
      }
#pragma unroll
      for (int j = 0; j < 4; ++j) {
        const int col = n0 + j * 16 + cl;
        const float bias = b3[col];
#pragma unroll
        for (int r = 0; r < 8; ++r)
          sh3[(r + hi) * N3 + col] = f2bf(selu_f(acc[j][r] + bias));
      }
    }
  }
  __syncthreads();

  // ---- Stage 4: recon = sigmoid(h3 @ W4^T + b4)  [16 x 264 (pad 272), K=1024] ----
  {
    auto store_recon = [&](const v8f& acc, int n0) {
      const int col = n0 + cl;
      if (col < N4) {
        const float bias = b4[col];
#pragma unroll
        for (int r = 0; r < 8; ++r) {
          float v = 1.f / (1.f + __expf(-(acc[r] + bias)));
          out[(size_t)(r0 + r + hi) * 264 + col] = v;
        }
      }
    };

    // waves 0..7 take tile pairs (2w, 2w+1) with a shared A fragment
    const int n0 = wave * 32;
    const unsigned short* wb0 = W4b + (size_t)(n0)      * N3;
    const unsigned short* wb1 = W4b + (size_t)(n0 + 16) * N3;
    v8f acc[2] = {};
    for (int kb = 0; kb < N3 / 32; ++kb) {
      const int k0 = kb * 32;
      __builtin_prefetch(wb0 + k0 + 64, 0, 3);
      __builtin_prefetch(wb1 + k0 + 64, 0, 3);
      v16bf a  = load_frag(sh3, N3, k0, lane);
      v16bf b0 = load_frag(wb0, N3, k0, lane);
      acc[0] = WMMA_BF16(a, b0, acc[0]);
      v16bf b1f = load_frag(wb1, N3, k0, lane);
      acc[1] = WMMA_BF16(a, b1f, acc[1]);
    }
    store_recon(acc[0], n0);
    store_recon(acc[1], n0 + 16);

    // masked tail tile (cols 256..263 valid) handled by wave 0
    if (wave == 0) {
      const unsigned short* wb = W4b + (size_t)256 * N3;
      v8f acct = {};
      for (int kb = 0; kb < N3 / 32; ++kb) {
        v16bf a = load_frag(sh3, N3, kb * 32, lane);
        v16bf b = load_frag(wb, N3, kb * 32, lane);
        acct = WMMA_BF16(a, b, acct);
      }
      store_recon(acct, 256);
    }
  }
}

extern "C" void kernel_launch(void* const* d_in, const int* in_sizes, int n_in,
                              void* d_out, int out_size, void* d_ws, size_t ws_size,
                              hipStream_t stream) {
  (void)in_sizes; (void)n_in; (void)out_size; (void)ws_size;
  const float* x   = (const float*)d_in[0];
  const float* eps = (const float*)d_in[1];
  const float* W1  = (const float*)d_in[2];
  const float* b1  = (const float*)d_in[3];
  const float* W21 = (const float*)d_in[4];
  const float* b21 = (const float*)d_in[5];
  const float* W22 = (const float*)d_in[6];
  const float* b22 = (const float*)d_in[7];
  const float* W3  = (const float*)d_in[8];
  const float* b3  = (const float*)d_in[9];
  const float* W4  = (const float*)d_in[10];
  const float* b4  = (const float*)d_in[11];
  float* out = (float*)d_out;

  // bf16 weight staging in workspace (ushort units)
  unsigned short* W1b  = (unsigned short*)d_ws;
  unsigned short* W21b = W1b  + (size_t)2048 * K1PAD;
  unsigned short* W22b = W21b + (size_t)128 * 2048;
  unsigned short* W3b  = W22b + (size_t)128 * 2048;
  unsigned short* W4b  = W3b  + (size_t)1024 * 128;

  cvt_pad<<<dim3(1024), dim3(256), 0, stream>>>(W1,  W1b,  2048, 528,  K1PAD, 2048);
  cvt_pad<<<dim3(256),  dim3(256), 0, stream>>>(W21, W21b, 128,  2048, 2048,  128);
  cvt_pad<<<dim3(256),  dim3(256), 0, stream>>>(W22, W22b, 128,  2048, 2048,  128);
  cvt_pad<<<dim3(128),  dim3(256), 0, stream>>>(W3,  W3b,  1024, 128,  128,   1024);
  cvt_pad<<<dim3(256),  dim3(256), 0, stream>>>(W4,  W4b,  264,  1024, 1024,  N4PAD);

  vae_fused<<<dim3(BTOT / 16), dim3(256), SMEM_BYTES, stream>>>(
      x, eps, W1b, b1, W21b, b21, W22b, b22, W3b, b3, W4b, b4, out);
}
